// LSTMEncoderAttnDecoder_68788196213303
// MI455X (gfx1250) — compile-verified, matrix operand order checked by reading, not run on hardware
//
#include <hip/hip_runtime.h>
#include <math.h>

typedef __bf16 bf16_t;
typedef __attribute__((ext_vector_type(16))) __bf16 v16bf;
typedef __attribute__((ext_vector_type(8)))  __bf16 v8bf;
typedef __attribute__((ext_vector_type(8)))  float  v8f;

namespace {
constexpr int Bc = 32, Sc = 256, Tc = 64, Ec = 256, Hc = 512, Vc = 32000;
constexpr int G4  = 4 * Hc;          // 2048 gate width
constexpr int Kd0 = Ec + Hc + Hc;    // 1280 decoder L0 input (emb|ctx|h_prev)
constexpr int Kd1 = Hc + Hc;         // 1024 decoder L1 input (h0_new|h1_prev)
constexpr float ATT_SCALE = 0.044194173824159216f; // 1/sqrt(512)
}

// ---------------------------------------------------------------------------
// bf16 fragment load matching CDNA5 16-bit A/B WMMA layout (16x32 per wave32).
// lane<16: row=base+lane,   K = {k0..k0+7, k0+16..k0+23}
// lane>=16: row=base+lane-16, K = {k0+8..k0+15, k0+24..k0+31}
// ---------------------------------------------------------------------------
__device__ __forceinline__ v16bf load_frag(const bf16_t* __restrict__ base, int ld,
                                           int row0, int k0, int lane) {
  const int r    = lane & 15;
  const int koff = (lane >> 4) << 3;             // 0 or 8
  const bf16_t* p = base + (size_t)(row0 + r) * ld + k0 + koff;
  v8bf lo = *(const v8bf*)(p);
  v8bf hi = *(const v8bf*)(p + 16);
  v16bf out;
#pragma unroll
  for (int i = 0; i < 8; ++i) { out[i] = lo[i]; out[i + 8] = hi[i]; }
  return out;
}

// C[M,N](f32) = A[M,K](bf16, row-major) x W[N,K](bf16, row-major)^T (+bias)
// block = 4 waves; wave w -> rows [by*64+16w, +16); cols [bx*64, +64)
__global__ __launch_bounds__(128) void wmma_gemm_bf16(
    const bf16_t* __restrict__ A, int lda,
    const bf16_t* __restrict__ W, int ldw,
    const float* __restrict__ bias,
    float* __restrict__ C, int ldc,
    int M, int N, int K) {
  const int lane = threadIdx.x & 31;
  const int wave = threadIdx.x >> 5;
  const int m0 = blockIdx.y * 64 + wave * 16;
  const int n0 = blockIdx.x * 64;
  if (m0 >= M) return;                                   // uniform per wave
  v8f a0 = {}, a1 = {}, a2 = {}, a3 = {};
  for (int k0 = 0; k0 < K; k0 += 32) {
    v16bf fa = load_frag(A, lda, m0, k0, lane);
    v16bf b0 = load_frag(W, ldw, n0 +  0, k0, lane);
    v16bf b1 = load_frag(W, ldw, n0 + 16, k0, lane);
    v16bf b2 = load_frag(W, ldw, n0 + 32, k0, lane);
    v16bf b3 = load_frag(W, ldw, n0 + 48, k0, lane);
    a0 = __builtin_amdgcn_wmma_f32_16x16x32_bf16(false, fa, false, b0, (short)0, a0, false, false);
    a1 = __builtin_amdgcn_wmma_f32_16x16x32_bf16(false, fa, false, b1, (short)0, a1, false, false);
    a2 = __builtin_amdgcn_wmma_f32_16x16x32_bf16(false, fa, false, b2, (short)0, a2, false, false);
    a3 = __builtin_amdgcn_wmma_f32_16x16x32_bf16(false, fa, false, b3, (short)0, a3, false, false);
  }
  // C/D layout: VGPR r -> row m0 + r + 8*(lane>=16); col = n0 + lane%16 (+16 per tile)
  const int row = m0 + ((lane >> 4) << 3);
  const int col = n0 + (lane & 15);
  v8f acc[4] = {a0, a1, a2, a3};
#pragma unroll
  for (int t = 0; t < 4; ++t) {
    const int c = col + 16 * t;
    const float bv = bias ? bias[c] : 0.0f;
#pragma unroll
    for (int r = 0; r < 8; ++r)
      C[(size_t)(row + r) * ldc + c] = acc[t][r] + bv;
  }
}

// ---------------------------------------------------------------------------
// Pointwise / utility kernels
// ---------------------------------------------------------------------------
__global__ void cast_bf16_kernel(const float* __restrict__ src, bf16_t* __restrict__ dst, size_t n) {
  size_t i = (size_t)blockIdx.x * blockDim.x + threadIdx.x;
  if (i < n) dst[i] = (bf16_t)src[i];
}

// dst[n, Ka+Kb] = [A[n,Ka] | B[n,Kb]]  (fp32 -> bf16)
__global__ void cat_cast_kernel(const float* __restrict__ A, int Ka,
                                const float* __restrict__ Bm, int Kb,
                                bf16_t* __restrict__ dst, int Nrows) {
  const int Kt = Ka + Kb;
  size_t i = (size_t)blockIdx.x * blockDim.x + threadIdx.x;
  if (i >= (size_t)Nrows * Kt) return;
  int k = (int)(i % Kt);
  int n = (int)(i / Kt);
  float v = (k < Ka) ? A[(size_t)n * Ka + k] : Bm[(size_t)n * Kb + (k - Ka)];
  dst[i] = (bf16_t)v;
}

__global__ void fill_zero_f32(float* p, int n) {
  int i = blockIdx.x * blockDim.x + threadIdx.x;
  if (i < n) p[i] = 0.0f;
}
__global__ void fill_zero_bf16(bf16_t* p, int n) {
  int i = blockIdx.x * blockDim.x + threadIdx.x;
  if (i < n) p[i] = (bf16_t)0.0f;
}

// out[(s*B+b)*E + e] = emb[idx[b,s]*E + e]  (fp32 -> bf16), time-major rows
__global__ void gather_embed_kernel(const int* __restrict__ idx, const float* __restrict__ emb,
                                    bf16_t* __restrict__ out, int Bn, int Sn, int En) {
  size_t i = (size_t)blockIdx.x * blockDim.x + threadIdx.x;
  if (i >= (size_t)Bn * Sn * En) return;
  int e = (int)(i % En);
  size_t r = i / En;
  int b = (int)(r % Bn);
  int s = (int)(r / Bn);
  int tok = idx[(size_t)b * Sn + s];
  out[i] = (bf16_t)emb[(size_t)tok * En + e];
}

__device__ __forceinline__ float sigm(float x) { return 1.0f / (1.0f + expf(-x)); }

// LSTM cell pointwise: gates = Gx (+ Gh) + bias ; PyTorch order i,f,g,o.
// Optional bf16 mirrors of h at dest + b*ld + j (up to 3) and an f32 archive.
__global__ void lstm_cell_kernel(
    const float* __restrict__ Gx, const float* __restrict__ Gh,
    const float* __restrict__ bias,
    float* __restrict__ c, float* __restrict__ h,
    bf16_t* hb0, int ld0, bf16_t* hb1, int ld1, bf16_t* hb2, int ld2,
    float* hf, int ldf) {
  int i = blockIdx.x * blockDim.x + threadIdx.x;
  if (i >= Bc * Hc) return;
  int b = i >> 9, j = i & (Hc - 1);
  size_t base = (size_t)b * G4;
  float gi = Gx[base + j]            + bias[j];
  float gf = Gx[base + Hc + j]       + bias[Hc + j];
  float gg = Gx[base + 2 * Hc + j]   + bias[2 * Hc + j];
  float go = Gx[base + 3 * Hc + j]   + bias[3 * Hc + j];
  if (Gh) {
    gi += Gh[base + j]; gf += Gh[base + Hc + j];
    gg += Gh[base + 2 * Hc + j]; go += Gh[base + 3 * Hc + j];
  }
  float cn = sigm(gf) * c[i] + sigm(gi) * tanhf(gg);
  float hn = sigm(go) * tanhf(cn);
  c[i] = cn; h[i] = hn;
  if (hb0) hb0[(size_t)b * ld0 + j] = (bf16_t)hn;
  if (hb1) hb1[(size_t)b * ld1 + j] = (bf16_t)hn;
  if (hb2) hb2[(size_t)b * ld2 + j] = (bf16_t)hn;
  if (hf)  hf[(size_t)b * ldf + j]  = hn;
}

// scores[b,s] = SCALE * sum_h tanh(q[b,h] + keys[(s*B+b),h]) * We[h], masked
__global__ void scores_kernel(const float* __restrict__ q, const float* __restrict__ keys,
                              const float* __restrict__ We, const int* __restrict__ mask,
                              float* __restrict__ scores) {
  int i = blockIdx.x * blockDim.x + threadIdx.x;
  if (i >= Bc * Sc) return;
  int b = i >> 8, s = i & (Sc - 1);
  const float* qr = q + (size_t)b * Hc;
  const float* kr = keys + ((size_t)s * Bc + b) * Hc;
  float acc = 0.0f;
  for (int hh = 0; hh < Hc; ++hh) acc += tanhf(qr[hh] + kr[hh]) * We[hh];
  acc *= ATT_SCALE;
  if (mask[(size_t)b * Sc + s] == 0) acc = -1e9f;
  scores[i] = acc;
}

// one block per batch row; also archives attn into [B,T,S] output
__global__ void softmax_kernel(const float* __restrict__ scores, float* __restrict__ attn,
                               float* __restrict__ attns_out, int t) {
  __shared__ float red[Sc];
  int b = blockIdx.x, s = threadIdx.x;
  float v = scores[(size_t)b * Sc + s];
  red[s] = v; __syncthreads();
  for (int off = Sc / 2; off; off >>= 1) { if (s < off) red[s] = fmaxf(red[s], red[s + off]); __syncthreads(); }
  float m = red[0]; __syncthreads();
  float e = expf(v - m);
  red[s] = e; __syncthreads();
  for (int off = Sc / 2; off; off >>= 1) { if (s < off) red[s] += red[s + off]; __syncthreads(); }
  float p = e / red[0];
  attn[(size_t)b * Sc + s] = p;
  attns_out[((size_t)b * Tc + t) * Sc + s] = p;  // [B,T,S]
}

// ctx[b,h] = sum_s attn[b,s] * enc_out_f32[(s*B+b),h]; mirrored to 2 bf16 dests
__global__ void ctx_kernel(const float* __restrict__ attn, const float* __restrict__ enc_out,
                           bf16_t* __restrict__ d0, int ld0, bf16_t* __restrict__ d1, int ld1) {
  int i = blockIdx.x * blockDim.x + threadIdx.x;
  if (i >= Bc * Hc) return;
  int b = i >> 9, j = i & (Hc - 1);
  float acc = 0.0f;
  for (int s = 0; s < Sc; ++s)
    acc += attn[(size_t)b * Sc + s] * enc_out[((size_t)s * Bc + b) * Hc + j];
  d0[(size_t)b * ld0 + j] = (bf16_t)acc;
  d1[(size_t)b * ld1 + j] = (bf16_t)acc;
}

// decoder input: cat0[b, 0..E) = demb_t[(b)*E + e]
__global__ void copy_emb_kernel(const bf16_t* __restrict__ demb_t, bf16_t* __restrict__ cat0) {
  int i = blockIdx.x * blockDim.x + threadIdx.x;
  if (i >= Bc * Ec) return;
  int b = i / Ec, e = i % Ec;
  cat0[(size_t)b * Kd0 + e] = demb_t[i];
}

// decoder initial state <- encoder final state; seed bf16 h slots
__global__ void seed_dec_kernel(const float* eh0, const float* ec0, const float* eh1, const float* ec1,
                                float* dh0, float* dc0, float* dh1, float* dc1,
                                bf16_t* cat0, bf16_t* cat1, bf16_t* htop) {
  int i = blockIdx.x * blockDim.x + threadIdx.x;
  if (i >= Bc * Hc) return;
  int b = i >> 9, j = i & (Hc - 1);
  dh0[i] = eh0[i]; dc0[i] = ec0[i]; dh1[i] = eh1[i]; dc1[i] = ec1[i];
  cat0[(size_t)b * Kd0 + (Ec + Hc) + j] = (bf16_t)eh0[i];  // h_prev L0
  cat1[(size_t)b * Kd1 + Hc + j]        = (bf16_t)eh1[i];  // h_prev L1
  htop[(size_t)b * Hc + j]              = (bf16_t)eh1[i];
}

__global__ void final_state_kernel(const float* dh0, const float* dc0,
                                   const float* dh1, const float* dc1,
                                   float* hid_o, float* cel_o) {
  int i = blockIdx.x * blockDim.x + threadIdx.x;
  if (i >= Bc * Hc) return;
  hid_o[i] = dh0[i]; hid_o[Bc * Hc + i] = dh1[i];
  cel_o[i] = dc0[i]; cel_o[Bc * Hc + i] = dc1[i];
}

// ---------------------------------------------------------------------------
// Host orchestration
// ---------------------------------------------------------------------------
extern "C" void kernel_launch(void* const* d_in, const int* in_sizes, int n_in,
                              void* d_out, int out_size, void* d_ws, size_t ws_size,
                              hipStream_t stream) {
  (void)in_sizes; (void)n_in; (void)out_size; (void)ws_size;
  const int*   enc_idx  = (const int*)d_in[0];
  const int*   dec_idx  = (const int*)d_in[1];
  const int*   enc_mask = (const int*)d_in[2];
  const float* emb_enc  = (const float*)d_in[3];
  const float* emb_dec  = (const float*)d_in[4];
  const float* eW0ih = (const float*)d_in[5];
  const float* eW0hh = (const float*)d_in[6];
  const float* eB0   = (const float*)d_in[7];
  const float* eW1ih = (const float*)d_in[8];
  const float* eW1hh = (const float*)d_in[9];
  const float* eB1   = (const float*)d_in[10];
  const float* dW0ih = (const float*)d_in[11];
  const float* dW0hh = (const float*)d_in[12];
  const float* dB0   = (const float*)d_in[13];
  const float* dW1ih = (const float*)d_in[14];
  const float* dW1hh = (const float*)d_in[15];
  const float* dB1   = (const float*)d_in[16];
  const float* Wq = (const float*)d_in[17];
  const float* Wk = (const float*)d_in[18];
  const float* We = (const float*)d_in[19];
  const float* Wo = (const float*)d_in[20];
  const float* bo = (const float*)d_in[21];

  float* logits = (float*)d_out;                           // [B,T,V]
  float* attns  = logits + (size_t)Bc * Tc * Vc;           // [B,T,S]
  float* hid_o  = attns + (size_t)Bc * Tc * Sc;            // [L,B,H]
  float* cel_o  = hid_o + (size_t)2 * Bc * Hc;             // [L,B,H]

  // workspace carve-out (256B aligned)
  size_t off = 0;
  auto alloc = [&](size_t bytes) -> void* {
    void* p = (char*)d_ws + off;
    off += (bytes + 255) & ~(size_t)255;
    return p;
  };
  bf16_t* X0     = (bf16_t*)alloc((size_t)Sc * Bc * Ec * 2);       // enc embeds, time-major
  bf16_t* demb   = (bf16_t*)alloc((size_t)Tc * Bc * Ec * 2);       // dec embeds
  bf16_t* W0ihb  = (bf16_t*)alloc((size_t)G4 * Ec * 2);
  bf16_t* W0hhb  = (bf16_t*)alloc((size_t)G4 * Hc * 2);
  bf16_t* W1ihb  = (bf16_t*)alloc((size_t)G4 * Hc * 2);
  bf16_t* W1hhb  = (bf16_t*)alloc((size_t)G4 * Hc * 2);
  bf16_t* Wcat0b = (bf16_t*)alloc((size_t)G4 * Kd0 * 2);           // [dW0ih | dW0hh]
  bf16_t* Wcat1b = (bf16_t*)alloc((size_t)G4 * Kd1 * 2);           // [dW1ih | dW1hh]
  bf16_t* Wqb    = (bf16_t*)alloc((size_t)Hc * Hc * 2);
  bf16_t* Wkb    = (bf16_t*)alloc((size_t)Hc * Hc * 2);
  bf16_t* Wob    = (bf16_t*)alloc((size_t)Vc * 2 * Hc * 2);
  float*  Xg     = (float*)alloc((size_t)Sc * Bc * G4 * 4);        // hoisted input gates
  bf16_t* xs1    = (bf16_t*)alloc((size_t)Sc * Bc * Hc * 2);       // enc L0 outputs
  bf16_t* xs2    = (bf16_t*)alloc((size_t)Sc * Bc * Hc * 2);       // enc L1 outputs (bf16)
  float*  encf   = (float*)alloc((size_t)Sc * Bc * Hc * 4);        // enc L1 outputs (f32)
  float*  keys   = (float*)alloc((size_t)Sc * Bc * Hc * 4);        // enc_out @ Wk^T
  float*  Hg     = (float*)alloc((size_t)Bc * G4 * 4);             // recurrent gates
  float*  Gg     = (float*)alloc((size_t)Bc * G4 * 4);             // decoder fused gates
  float*  eh0 = (float*)alloc(Bc * Hc * 4); float* ec0 = (float*)alloc(Bc * Hc * 4);
  float*  eh1 = (float*)alloc(Bc * Hc * 4); float* ec1 = (float*)alloc(Bc * Hc * 4);
  bf16_t* eh0b = (bf16_t*)alloc(Bc * Hc * 2); bf16_t* eh1b = (bf16_t*)alloc(Bc * Hc * 2);
  float*  dh0 = (float*)alloc(Bc * Hc * 4); float* dc0 = (float*)alloc(Bc * Hc * 4);
  float*  dh1 = (float*)alloc(Bc * Hc * 4); float* dc1 = (float*)alloc(Bc * Hc * 4);
  bf16_t* cat0 = (bf16_t*)alloc((size_t)Bc * Kd0 * 2);
  bf16_t* cat1 = (bf16_t*)alloc((size_t)Bc * Kd1 * 2);
  bf16_t* htop = (bf16_t*)alloc(Bc * Hc * 2);
  float*  qbuf = (float*)alloc(Bc * Hc * 4);
  float*  scrs = (float*)alloc(Bc * Sc * 4);
  float*  attn = (float*)alloc(Bc * Sc * 4);
  bf16_t* outs = (bf16_t*)alloc((size_t)Bc * Tc * 2 * Hc * 2);     // rows b*T+t

  auto gemm = [&](const bf16_t* A, int lda, const bf16_t* W, int ldw, const float* bias,
                  float* C, int ldc, int M, int N, int K) {
    dim3 g(N / 64, (M + 63) / 64);
    wmma_gemm_bf16<<<g, dim3(128), 0, stream>>>(A, lda, W, ldw, bias, C, ldc, M, N, K);
  };
  auto cast = [&](const float* s, bf16_t* d, size_t n) {
    cast_bf16_kernel<<<(unsigned)((n + 255) / 256), 256, 0, stream>>>(s, d, n);
  };

  // ---- weight conversion / packing ----
  cast(eW0ih, W0ihb, (size_t)G4 * Ec);
  cast(eW0hh, W0hhb, (size_t)G4 * Hc);
  cast(eW1ih, W1ihb, (size_t)G4 * Hc);
  cast(eW1hh, W1hhb, (size_t)G4 * Hc);
  cast(Wq, Wqb, (size_t)Hc * Hc);
  cast(Wk, Wkb, (size_t)Hc * Hc);
  cast(Wo, Wob, (size_t)Vc * 2 * Hc);
  {
    size_t n0 = (size_t)G4 * Kd0;
    cat_cast_kernel<<<(unsigned)((n0 + 255) / 256), 256, 0, stream>>>(dW0ih, Ec + Hc, dW0hh, Hc, Wcat0b, G4);
    size_t n1 = (size_t)G4 * Kd1;
    cat_cast_kernel<<<(unsigned)((n1 + 255) / 256), 256, 0, stream>>>(dW1ih, Hc, dW1hh, Hc, Wcat1b, G4);
  }

  // ---- embeddings ----
  gather_embed_kernel<<<(Bc * Sc * Ec + 255) / 256, 256, 0, stream>>>(enc_idx, emb_enc, X0, Bc, Sc, Ec);
  gather_embed_kernel<<<(Bc * Tc * Ec + 255) / 256, 256, 0, stream>>>(dec_idx, emb_dec, demb, Bc, Tc, Ec);

  // ---- encoder state init ----
  fill_zero_f32<<<(Bc * Hc + 255) / 256, 256, 0, stream>>>(eh0, Bc * Hc);
  fill_zero_f32<<<(Bc * Hc + 255) / 256, 256, 0, stream>>>(ec0, Bc * Hc);
  fill_zero_f32<<<(Bc * Hc + 255) / 256, 256, 0, stream>>>(eh1, Bc * Hc);
  fill_zero_f32<<<(Bc * Hc + 255) / 256, 256, 0, stream>>>(ec1, Bc * Hc);
  fill_zero_bf16<<<(Bc * Hc + 255) / 256, 256, 0, stream>>>(eh0b, Bc * Hc);
  fill_zero_bf16<<<(Bc * Hc + 255) / 256, 256, 0, stream>>>(eh1b, Bc * Hc);

  const dim3 cellG((Bc * Hc + 255) / 256), cellB(256);

  // ---- encoder layer 0: hoisted input gates, then recurrence ----
  gemm(X0, Ec, W0ihb, Ec, nullptr, Xg, G4, Sc * Bc, G4, Ec);
  for (int t = 0; t < Sc; ++t) {
    gemm(eh0b, Hc, W0hhb, Hc, nullptr, Hg, G4, Bc, G4, Hc);
    lstm_cell_kernel<<<cellG, cellB, 0, stream>>>(
        Xg + (size_t)t * Bc * G4, Hg, eB0, ec0, eh0,
        eh0b, Hc, xs1 + (size_t)t * Bc * Hc, Hc, nullptr, 0, nullptr, 0);
  }
  // ---- encoder layer 1 ----
  gemm(xs1, Hc, W1ihb, Hc, nullptr, Xg, G4, Sc * Bc, G4, Hc);
  for (int t = 0; t < Sc; ++t) {
    gemm(eh1b, Hc, W1hhb, Hc, nullptr, Hg, G4, Bc, G4, Hc);
    lstm_cell_kernel<<<cellG, cellB, 0, stream>>>(
        Xg + (size_t)t * Bc * G4, Hg, eB1, ec1, eh1,
        eh1b, Hc, xs2 + (size_t)t * Bc * Hc, Hc, nullptr, 0,
        encf + (size_t)t * Bc * Hc, Hc);
  }

  // ---- attention keys (hoisted) + decoder state seeding ----
  gemm(xs2, Hc, Wkb, Hc, nullptr, keys, Hc, Sc * Bc, Hc, Hc);
  seed_dec_kernel<<<cellG, cellB, 0, stream>>>(eh0, ec0, eh1, ec1, dh0, dc0, dh1, dc1, cat0, cat1, htop);

  // ---- decoder: T steps of attention + 2 fused LSTM layers ----
  for (int t = 0; t < Tc; ++t) {
    gemm(htop, Hc, Wqb, Hc, nullptr, qbuf, Hc, Bc, Hc, Hc);               // q = h_top @ Wq^T
    scores_kernel<<<(Bc * Sc + 255) / 256, 256, 0, stream>>>(qbuf, keys, We, enc_mask, scrs);
    softmax_kernel<<<Bc, Sc, 0, stream>>>(scrs, attn, attns, t);
    ctx_kernel<<<cellG, cellB, 0, stream>>>(attn, encf,
        cat0 + Ec, Kd0, outs + (size_t)t * 2 * Hc + Hc, Tc * 2 * Hc);     // ctx -> cat0 & outs
    copy_emb_kernel<<<(Bc * Ec + 255) / 256, 256, 0, stream>>>(demb + (size_t)t * Bc * Ec, cat0);
    // layer 0: fused [emb|ctx|h_prev] @ [Wih|Whh]^T
    gemm(cat0, Kd0, Wcat0b, Kd0, nullptr, Gg, G4, Bc, G4, Kd0);
    lstm_cell_kernel<<<cellG, cellB, 0, stream>>>(
        Gg, nullptr, dB0, dc0, dh0,
        cat0 + (Ec + Hc), Kd0, cat1, Kd1, nullptr, 0, nullptr, 0);
    // layer 1: fused [h0_new|h1_prev] @ [Wih|Whh]^T
    gemm(cat1, Kd1, Wcat1b, Kd1, nullptr, Gg, G4, Bc, G4, Kd1);
    lstm_cell_kernel<<<cellG, cellB, 0, stream>>>(
        Gg, nullptr, dB1, dc1, dh1,
        cat1 + Hc, Kd1, htop, Hc, outs + (size_t)t * 2 * Hc, Tc * 2 * Hc, nullptr, 0);
  }

  // ---- output projection: [B*T, 2H] x Wo[V, 2H]^T + bo -> logits [B,T,V] ----
  gemm(outs, 2 * Hc, Wob, 2 * Hc, bo, logits, Vc, Bc * Tc, Vc, 2 * Hc);

  final_state_kernel<<<cellG, cellB, 0, stream>>>(dh0, dc0, dh1, dc1, hid_o, cel_o);
}
